// BlockSelfAttention_2508260901599
// MI455X (gfx1250) — matrix operand order repaired and checked by reading, not compiled, not run on hardware
//
#include <hip/hip_runtime.h>
#include <cstdint>
#include <cstddef>

// ---------------------------------------------------------------------------
// Types for CDNA5 WMMA (wave32): A/B are 16 bf16 per lane, C/D are 8 f32.
// ---------------------------------------------------------------------------
typedef __bf16 bf16_t;
typedef __attribute__((ext_vector_type(16))) __bf16 v16bf;
typedef __attribute__((ext_vector_type(8)))  __bf16 v8bf;
typedef __attribute__((ext_vector_type(4)))  __bf16 v4bf;
typedef __attribute__((ext_vector_type(8)))  float  v8f;
typedef unsigned int u32t;
typedef __attribute__((ext_vector_type(4))) unsigned int v4u;
typedef __attribute__((ext_vector_type(8))) int v8i;
typedef __attribute__((ext_vector_type(4))) int v4i;

static __device__ __forceinline__ v8f wmma_bf16(v16bf a, v16bf b, v8f c) {
  return __builtin_amdgcn_wmma_f32_16x16x32_bf16(false, a, false, b,
                                                 (short)0, c, false, false);
}

static __device__ __forceinline__ v16bf make_frag(const bf16_t* p0, const bf16_t* p1) {
  v8bf a = *(const v8bf*)p0;
  v8bf b = *(const v8bf*)p1;
  return __builtin_shufflevector(a, b, 0,1,2,3,4,5,6,7,8,9,10,11,12,13,14,15);
}

// ---------------------------------------------------------------------------
// DPP16 butterfly reduction across the 16 lanes of a half-wave (no LDS).
// quad_perm(1,0,3,2)=0xB1, quad_perm(2,3,0,1)=0x4E, row_ror:4=0x124, row_ror:8=0x128
// ---------------------------------------------------------------------------
template<int CTRL>
static __device__ __forceinline__ float dpp_mov(float x) {
  return __builtin_bit_cast(float,
      __builtin_amdgcn_mov_dpp(__builtin_bit_cast(int, x), CTRL, 0xF, 0xF, true));
}
static __device__ __forceinline__ float red_max16(float x) {
  x = fmaxf(x, dpp_mov<0x0B1>(x));
  x = fmaxf(x, dpp_mov<0x04E>(x));
  x = fmaxf(x, dpp_mov<0x124>(x));
  x = fmaxf(x, dpp_mov<0x128>(x));
  return x;
}
static __device__ __forceinline__ float red_sum16(float x) {
  x += dpp_mov<0x0B1>(x);
  x += dpp_mov<0x04E>(x);
  x += dpp_mov<0x124>(x);
  x += dpp_mov<0x128>(x);
  return x;
}

// ---------------------------------------------------------------------------
// Tensor Data Mover: 2D tile (bf16) global -> LDS with hardware row padding.
// D# packed per CDNA5 ISA 08_async_tensor §8 (group0: count/lds/global/type,
// group1: data_size=2B, pad cfg, tensor dims, tile dims, dim0 stride).
// This toolchain exposes the 6-argument builtin:
//   (uint32x4 g0, int32x8 g1, int32x4 g2, int32x4 g3, int32x8 g4, i32 cpol)
// ---------------------------------------------------------------------------
#if __has_builtin(__builtin_amdgcn_tensor_load_to_lds)
#define USE_TDM 1
#endif

#ifdef USE_TDM
static __device__ __forceinline__ u32t lds_off_u32(const void* p) {
  return (u32t)(uintptr_t)p;   // generic LDS pointer: addr[31:0] = LDS offset
}
static __device__ __forceinline__ void tdm_load_2d(
    u32t lds_addr, const bf16_t* gptr, u32t stride_d0_elems,
    u32t tile_d0, u32t tile_d1, u32t pad_int_code, u32t pad_amt_code)
{
  const unsigned long long ga = (unsigned long long)(uintptr_t)gptr;
  v4u g0;
  g0[0] = 1u;                                        // count=1, user descriptor
  g0[1] = lds_addr;                                  // lds_addr (bytes)
  g0[2] = (u32t)ga;                                  // global_addr[31:0]
  g0[3] = (u32t)((ga >> 32) & 0x01ffffffu)           // global_addr[56:32]
        | 0x80000000u;                               // type=2 ("image") @[127:126]
  const u32t td0 = 1u << 20, td1 = 1u << 20;         // huge tensor dims (in-bounds tiles)
  v8i g1;
  g1[0] = (int)((1u << 16)                           // data_size = 2 bytes
              | (1u << 20)                           // pad_enable
              | (pad_int_code << 22)                 // pad interval (DWORDs, coded)
              | (pad_amt_code << 25));               // pad amount (DWORDs, coded)
  g1[1] = (int)((td0 & 0xffffu) << 16);              // tensor_dim0[15:0]
  g1[2] = (int)(((td0 >> 16) & 0xffffu) | ((td1 & 0xffffu) << 16));
  g1[3] = (int)(((td1 >> 16) & 0xffffu) | ((tile_d0 & 0xffffu) << 16));
  g1[4] = (int)(tile_d1 & 0xffffu);                  // tile_dim1 (tile_dim2 = 0)
  g1[5] = (int)stride_d0_elems;                      // tensor_dim0_stride[31:0]
  g1[6] = 0;
  g1[7] = 0;
  const v4i z4 = {0, 0, 0, 0};
  const v8i z8 = {0, 0, 0, 0, 0, 0, 0, 0};
  __builtin_amdgcn_tensor_load_to_lds(g0, g1, z4, z4, z8, 0);
}
#endif

// ---------------------------------------------------------------------------
// f32 -> bf16 bulk conversion (x and weights), 4 elements/thread.
// ---------------------------------------------------------------------------
__global__ void __launch_bounds__(256) cvt_bf16(const float* __restrict__ in,
                                                bf16_t* __restrict__ out, int n) {
  const int i = (blockIdx.x * 256 + threadIdx.x) * 4;
  if (i < n) {
    const float4 f = *(const float4*)(in + i);
    v4bf o = {(bf16_t)f.x, (bf16_t)f.y, (bf16_t)f.z, (bf16_t)f.w};
    *(v4bf*)(out + i) = o;
  }
}

// ---------------------------------------------------------------------------
// Tiled GEMM:  C[M,N] = A[M,K](bf16) x W[N,K](bf16)^T  (+bias)(+resid)(gelu)
// 128x128 tile per 256-thread WG, 8 waves (2x4), K-step 32, TDM double-buffer.
// EPI: 0 = bias -> bf16;  1 = bias+resid -> f32;  2 = bias+gelu -> bf16
// ---------------------------------------------------------------------------
template<int EPI>
__global__ void __launch_bounds__(256) gemm_wmma(
    const bf16_t* __restrict__ A, const bf16_t* __restrict__ W,
    const float* __restrict__ bias, const float* __restrict__ resid,
    void* __restrict__ Cout, int M, int N, int K)
{
  constexpr int LSTR = 40;                 // 64B row + 16B pad = 80B stride
  __shared__ bf16_t As[2][128 * LSTR];
  __shared__ bf16_t Bs[2][128 * LSTR];

  const int t    = threadIdx.x;
  const int lane = t & 31;
  const int l    = lane & 15, hi = lane >> 4;
  const int wid  = __builtin_amdgcn_readfirstlane(t >> 5);
  const int waveM = wid >> 2, waveN = wid & 3;
  const int bm = blockIdx.y * 128;
  const int bn = blockIdx.x * 128;

  const v8f vzero = {0.f,0.f,0.f,0.f,0.f,0.f,0.f,0.f};
  v8f acc[4][2];
  for (int i = 0; i < 4; ++i)
    for (int j = 0; j < 2; ++j) acc[i][j] = vzero;

  const int nk = K >> 5;

#ifdef USE_TDM
  // Each wave DMAs its own 16-row slice of the A and B tiles.
  auto issue = [&](int i) {
    const int buf = i & 1;
    tdm_load_2d(lds_off_u32(&As[buf][wid * 16 * LSTR]),
                A + (size_t)(bm + wid * 16) * K + i * 32,
                (u32t)K, 32u, 16u, 3u, 3u);          // 16 DW rows, 4 DW pad
    tdm_load_2d(lds_off_u32(&Bs[buf][wid * 16 * LSTR]),
                W + (size_t)(bn + wid * 16) * K + i * 32,
                (u32t)K, 32u, 16u, 3u, 3u);
  };
  issue(0);
#else
  const int lr = t >> 1, lk = (t & 1) * 16;
#endif

  for (int i = 0; i < nk; ++i) {
#ifdef USE_TDM
    if (i + 1 < nk) {
      __syncthreads();                 // everyone done reading buf[(i+1)&1]
      issue(i + 1);
      __builtin_amdgcn_s_wait_tensorcnt(2);   // stage i's pair complete
    } else {
      __builtin_amdgcn_s_wait_tensorcnt(0);
    }
    __syncthreads();                   // publish stage i
    const bf16_t* as = As[i & 1];
    const bf16_t* bs = Bs[i & 1];
#else
    __syncthreads();
    {
      const bf16_t* srcA = A + (size_t)(bm + lr) * K + i * 32 + lk;
      bf16_t* dst = &As[0][lr * LSTR + lk];
      *(v8bf*)dst       = *(const v8bf*)srcA;
      *(v8bf*)(dst + 8) = *(const v8bf*)(srcA + 8);
      const bf16_t* srcB = W + (size_t)(bn + lr) * K + i * 32 + lk;
      bf16_t* dstB = &Bs[0][lr * LSTR + lk];
      *(v8bf*)dstB       = *(const v8bf*)srcB;
      *(v8bf*)(dstB + 8) = *(const v8bf*)(srcB + 8);
    }
    __syncthreads();
    const bf16_t* as = As[0];
    const bf16_t* bs = Bs[0];
#endif

    v16bf af[4], bfr[2];
    for (int mt = 0; mt < 4; ++mt) {
      const bf16_t* ar = &as[(waveM * 64 + mt * 16 + l) * LSTR];
      af[mt] = make_frag(ar + hi * 8, ar + 16 + hi * 8);       // A 16x32 layout
    }
    for (int nt = 0; nt < 2; ++nt) {
      const bf16_t* br = &bs[(waveN * 32 + nt * 16 + l) * LSTR];
      bfr[nt] = make_frag(br + hi * 8, br + 16 + hi * 8);      // B 32x16 layout
    }
    for (int mt = 0; mt < 4; ++mt)
      for (int nt = 0; nt < 2; ++nt)
        acc[mt][nt] = wmma_bf16(af[mt], bfr[nt], acc[mt][nt]);
  }

  // Epilogue. C layout: lane l/hi, VGPR v -> row = 16*tile + hi*8 + v, col = l.
  for (int mt = 0; mt < 4; ++mt) {
    for (int nt = 0; nt < 2; ++nt) {
      const int gc = bn + waveN * 32 + nt * 16 + l;
      const float bv = bias[gc];
      for (int v = 0; v < 8; ++v) {
        const int gr = bm + waveM * 64 + mt * 16 + hi * 8 + v;
        float val = acc[mt][nt][v] + bv;
        if constexpr (EPI == 1) {
          val += resid[(size_t)gr * N + gc];
          ((float*)Cout)[(size_t)gr * N + gc] = val;
        } else if constexpr (EPI == 2) {
          val = 0.5f * val * (1.0f + erff(val * 0.70710678118654752f));
          ((bf16_t*)Cout)[(size_t)gr * N + gc] = (bf16_t)val;
        } else {
          ((bf16_t*)Cout)[(size_t)gr * N + gc] = (bf16_t)val;
        }
      }
    }
  }
}

// ---------------------------------------------------------------------------
// Fused flash-style block attention. One WG per (block, head, 128-row strip).
// Each of 8 waves owns 16 query rows; online softmax in WMMA C-layout with
// DPP cross-lane reductions (no LDS traffic in the softmax).
// ---------------------------------------------------------------------------
__global__ void __launch_bounds__(256) attn_kernel(const bf16_t* __restrict__ qkv,
                                                   bf16_t* __restrict__ ctx)
{
  constexpr int KSTR = 72;
  constexpr int VSTR = 136;
  constexpr int PSTR = 40;
  __shared__ bf16_t Kl[128 * KSTR];
  __shared__ bf16_t Vt[64 * VSTR];
  __shared__ bf16_t Pst[8][16 * PSTR];

  const int bid   = blockIdx.x;
  const int strip = bid & 1;
  const int head  = (bid >> 1) & 7;
  const int blk   = bid >> 4;
  const size_t tb = (size_t)blk * 256;

  const bf16_t* Qp = qkv + tb * 1536 + head * 64;
  const bf16_t* Kp = Qp + 512;
  const bf16_t* Vp = Qp + 1024;

  const int t    = threadIdx.x;
  const int lane = t & 31, wid = t >> 5;
  const int l    = lane & 15, hi = lane >> 4;

  const int qrow = strip * 128 + wid * 16 + l;
  const bf16_t* qr = Qp + (size_t)qrow * 1536;
  const v16bf aq0 = make_frag(qr + hi * 16,      qr + hi * 16 + 8);
  const v16bf aq1 = make_frag(qr + 32 + hi * 16, qr + 32 + hi * 16 + 8);

  const v8f vzero = {0.f,0.f,0.f,0.f,0.f,0.f,0.f,0.f};
  float mrow[8], ssum[8];
  v8f cacc[4];
  for (int v = 0; v < 8; ++v) { mrow[v] = -1e30f; ssum[v] = 0.f; }
  for (int n = 0; n < 4; ++n) cacc[n] = vzero;

  const int lr = t >> 1;
  const int lc = (t & 1) * 32;

  for (int half = 0; half < 2; ++half) {
    __syncthreads();
    {   // stage K rows [half*128, +128) row-major
      const bf16_t* src = Kp + (size_t)(half * 128 + lr) * 1536 + lc;
      bf16_t* dst = &Kl[lr * KSTR + lc];
      for (int i = 0; i < 4; ++i)
        *(v8bf*)(dst + i * 8) = *(const v8bf*)(src + i * 8);
    }
    {   // stage V transposed: Vt[hd_col][kv_row_local]
      const bf16_t* src = Vp + (size_t)(half * 128 + lr) * 1536;
      for (int c = 0; c < 32; ++c)
        Vt[(lc + c) * VSTR + lr] = src[lc + c];
    }
    __syncthreads();

    for (int j = 0; j < 4; ++j) {        // 32 KV columns per step
      const int c0 = j * 32;
      v8f s0 = vzero, s1 = vzero;
      {
        const bf16_t* k0 = &Kl[(c0 + l) * KSTR];
        const bf16_t* k1 = &Kl[(c0 + 16 + l) * KSTR];
        v16bf b00 = make_frag(k0 + hi * 16,      k0 + hi * 16 + 8);
        v16bf b10 = make_frag(k1 + hi * 16,      k1 + hi * 16 + 8);
        s0 = wmma_bf16(aq0, b00, s0);
        s1 = wmma_bf16(aq0, b10, s1);
        v16bf b01 = make_frag(k0 + 32 + hi * 16, k0 + 32 + hi * 16 + 8);
        v16bf b11 = make_frag(k1 + 32 + hi * 16, k1 + 32 + hi * 16 + 8);
        s0 = wmma_bf16(aq1, b01, s0);
        s1 = wmma_bf16(aq1, b11, s1);
      }
      float f[8];
      for (int v = 0; v < 8; ++v) {
        const float a0 = s0[v] * 0.125f;         // 1/sqrt(64)
        const float a1 = s1[v] * 0.125f;
        const float tm = red_max16(fmaxf(a0, a1));
        const float mn = fmaxf(mrow[v], tm);
        f[v] = __expf(mrow[v] - mn);
        mrow[v] = mn;
        const float p0 = __expf(a0 - mn), p1 = __expf(a1 - mn);
        ssum[v] = ssum[v] * f[v] + red_sum16(p0 + p1);
        s0[v] = p0; s1[v] = p1;
      }
      for (int n = 0; n < 4; ++n)
        for (int v = 0; v < 8; ++v) cacc[n][v] *= f[v];

      // Re-layout P (C-layout f32 -> A-layout bf16) through per-wave staging.
      bf16_t* ps = &Pst[wid][0];
      for (int v = 0; v < 8; ++v) {
        ps[(v + hi * 8) * PSTR + l]      = (bf16_t)s0[v];
        ps[(v + hi * 8) * PSTR + 16 + l] = (bf16_t)s1[v];
      }
      __builtin_amdgcn_wave_barrier();   // same-wave LDS ops stay in order
      const v16bf ap = make_frag(&ps[l * PSTR + hi * 16],
                                 &ps[l * PSTR + hi * 16 + 8]);
      for (int n = 0; n < 4; ++n) {
        const bf16_t* vr = &Vt[(n * 16 + l) * VSTR + c0 + hi * 16];
        v16bf bv = make_frag(vr, vr + 8);
        cacc[n] = wmma_bf16(ap, bv, cacc[n]);
      }
    }
  }

  for (int n = 0; n < 4; ++n) {
    for (int v = 0; v < 8; ++v) {
      const float val = cacc[n][v] / ssum[v];
      const size_t row = tb + (size_t)(strip * 128 + wid * 16 + hi * 8 + v);
      const int col = head * 64 + n * 16 + l;
      ctx[row * 512 + col] = (bf16_t)val;
    }
  }
}

// ---------------------------------------------------------------------------
// Row LayerNorm over D=512. One 256-thread WG per row; optional bf16 mirror.
// ---------------------------------------------------------------------------
__global__ void __launch_bounds__(256) ln_kernel(const float* __restrict__ y,
                                                 const float* __restrict__ g,
                                                 const float* __restrict__ be,
                                                 float* __restrict__ of,
                                                 bf16_t* __restrict__ ob)
{
  const int row = blockIdx.x, t = threadIdx.x;
  const int lane = t & 31, wid = t >> 5;
  const float2 v = *(const float2*)(y + (size_t)row * 512 + t * 2);
  float s = v.x + v.y;
  float q = v.x * v.x + v.y * v.y;
  for (int m = 1; m < 32; m <<= 1) {
    s += __shfl_xor(s, m, 32);
    q += __shfl_xor(q, m, 32);
  }
  __shared__ float sb[8], qb[8];
  if (lane == 0) { sb[wid] = s; qb[wid] = q; }
  __syncthreads();
  s = 0.f; q = 0.f;
  for (int i = 0; i < 8; ++i) { s += sb[i]; q += qb[i]; }
  const float mu  = s * (1.f / 512.f);
  const float var = q * (1.f / 512.f) - mu * mu;
  const float rs  = rsqrtf(var + 1e-5f);
  const float o0 = (v.x - mu) * rs * g[t * 2]     + be[t * 2];
  const float o1 = (v.y - mu) * rs * g[t * 2 + 1] + be[t * 2 + 1];
  float2 o; o.x = o0; o.y = o1;
  *(float2*)(of + (size_t)row * 512 + t * 2) = o;
  if (ob) {
    ob[(size_t)row * 512 + t * 2]     = (bf16_t)o0;
    ob[(size_t)row * 512 + t * 2 + 1] = (bf16_t)o1;
  }
}

// ---------------------------------------------------------------------------
// Host side: conversions + 7 compute launches, scratch carved out of d_ws.
// ---------------------------------------------------------------------------
extern "C" void kernel_launch(void* const* d_in, const int* in_sizes, int n_in,
                              void* d_out, int out_size, void* d_ws, size_t ws_size,
                              hipStream_t stream) {
  (void)in_sizes; (void)n_in; (void)out_size; (void)ws_size;
  const float* x     = (const float*)d_in[0];
  const float* w_qkv = (const float*)d_in[1];
  const float* b_qkv = (const float*)d_in[2];
  const float* w_out = (const float*)d_in[3];
  const float* b_out = (const float*)d_in[4];
  const float* w1    = (const float*)d_in[5];
  const float* b1    = (const float*)d_in[6];
  const float* w2    = (const float*)d_in[7];
  const float* b2    = (const float*)d_in[8];
  const float* g1    = (const float*)d_in[9];
  const float* be1   = (const float*)d_in[10];
  const float* g2    = (const float*)d_in[11];
  const float* be2   = (const float*)d_in[12];
  float* out = (float*)d_out;

  char* ws = (char*)d_ws;
  bf16_t* qkv = (bf16_t*)(ws + 0);                 // [N,1536] bf16 (later: ff)
  bf16_t* ctx = (bf16_t*)(ws + 402653184ull);      // [N, 512] bf16
  float*  y1  = (float*)(ws + 536870912ull);       // [N, 512] f32 (-> h in place)
  bf16_t* hb  = (bf16_t*)(ws + 805306368ull);      // [N, 512] bf16
  float*  y2  = (float*)(ws + 939524096ull);       // [N, 512] f32 (early: xb)
  bf16_t* xb  = (bf16_t*)(ws + 939524096ull);      // [N, 512] bf16 (dead before y2)
  bf16_t* ffb = (bf16_t*)(ws + 0);                 // [N,1024] bf16 (reuses qkv)
  bf16_t* wqb = (bf16_t*)(ws + 1207959552ull);     // [1536,512] bf16
  bf16_t* wob = (bf16_t*)(ws + 1209532416ull);     // [ 512,512] bf16
  bf16_t* w1b = (bf16_t*)(ws + 1210056704ull);     // [1024,512] bf16
  bf16_t* w2b = (bf16_t*)(ws + 1211105280ull);     // [ 512,1024] bf16

  const int M = 131072;
  dim3 blk(256);

  // 0. One-shot f32 -> bf16 conversions (x + all GEMM weights).
  cvt_bf16<<<65536, blk, 0, stream>>>(x,     xb,  67108864);
  cvt_bf16<<<768,   blk, 0, stream>>>(w_qkv, wqb, 786432);
  cvt_bf16<<<256,   blk, 0, stream>>>(w_out, wob, 262144);
  cvt_bf16<<<512,   blk, 0, stream>>>(w1,    w1b, 524288);
  cvt_bf16<<<512,   blk, 0, stream>>>(w2,    w2b, 524288);

  // 1. QKV = X * Wqkv^T + b            -> bf16
  gemm_wmma<0><<<dim3(12, 1024), blk, 0, stream>>>(
      xb, wqb, b_qkv, nullptr, qkv, M, 1536, 512);
  // 2. Fused per-block attention       -> ctx bf16
  attn_kernel<<<8192, blk, 0, stream>>>(qkv, ctx);
  // 3. y1 = ctx * Wout^T + b + x       -> f32
  gemm_wmma<1><<<dim3(4, 1024), blk, 0, stream>>>(
      ctx, wob, b_out, x, y1, M, 512, 512);
  // 4. h = LN1(y1)  (in place f32 + bf16 mirror)
  ln_kernel<<<131072, blk, 0, stream>>>(y1, g1, be1, y1, hb);
  // 5. ff = gelu(h * W1^T + b)         -> bf16
  gemm_wmma<2><<<dim3(8, 1024), blk, 0, stream>>>(
      hb, w1b, b1, nullptr, ffb, M, 1024, 512);
  // 6. y2 = ff * W2^T + b + h          -> f32
  gemm_wmma<1><<<dim3(4, 1024), blk, 0, stream>>>(
      ffb, w2b, b2, y1, y2, M, 512, 1024);
  // 7. out = LN2(y2)
  ln_kernel<<<131072, blk, 0, stream>>>(y2, g2, be2, out, nullptr);
}
